// NoFoDifformer_org_7670811591173
// MI455X (gfx1250) — matrix operand order repaired and checked by reading, not compiled, not run on hardware
//
#include <hip/hip_runtime.h>
#include <math.h>

// ---------------- types ----------------
typedef __bf16 bf16_t;
typedef __attribute__((ext_vector_type(8)))  __bf16 v8bf;
typedef __attribute__((ext_vector_type(16))) __bf16 v16bf;
typedef __attribute__((ext_vector_type(8)))  float  v8f;

// problem constants
static constexpr int NN    = 8192;
static constexpr int NFEAT = 512;
static constexpr int HID   = 256;
static constexpr int CC    = 128;
static constexpr int KPOW  = 10;
static constexpr int DIMH  = 16;   // DIM/2

#define USE_ASYNC_LDS 1

__device__ __forceinline__ float gelu_exact(float x) {
    return 0.5f * x * (1.0f + erff(x * 0.70710678118654752f));
}

#if USE_ASYNC_LDS
// GLOBAL_LOAD_ASYNC_TO_LDS_B128: VDST = LDS byte offset (low 32 bits of generic
// shared pointer), VADDR = 64-bit global address. Tracked by ASYNCcnt.
__device__ __forceinline__ void async_load_b128(const void* gp, void* lp) {
    asm volatile("global_load_async_to_lds_b128 %0, %1, off"
                 :: "v"((unsigned)(unsigned long long)lp), "v"(gp)
                 : "memory");
}
__device__ __forceinline__ void wait_async0() {
    asm volatile("s_wait_asynccnt 0x0" ::: "memory");
}
#endif

// ---------------- WMMA GEMM ----------------
// C[M,N] = act(A @ B + bias), fp32 accumulate via v_wmma_f32_16x16x32_bf16.
// A: bf16 row-major [M,K]              (AF32=false)  -> async global->LDS
//    f32  row-major [M,K]              (AF32=true,  TRANSA=false) -> cvt on stage
//    f32  logical A[m][k]=mem[k*lda+m] (AF32=true,  TRANSA=true)  -> cvt+transpose
// Bt: bf16 [N,K] (B^T row-major) -> async global->LDS, contiguous b128.
// Tile 64x128, BK=32, 256 thr = 8 waves, each wave 2m x 2n WMMA tiles.
// Double-buffered LDS, one barrier per k-step.
template <int ACT, bool TRANSA, bool AF32, bool ATOMIC>
__global__ __launch_bounds__(256)
void gemm_wmma(const void* __restrict__ Av, const bf16_t* __restrict__ Bt,
               const float* __restrict__ bias, float* __restrict__ Cf,
               bf16_t* __restrict__ Cbf, int M, int N, int K, int lda) {
    __shared__ bf16_t As[2 * 64 * 40];   // row stride 40 halfs = 80B (16B aligned)
    __shared__ bf16_t Bs[2 * 128 * 40];

    const int tid  = threadIdx.x;
    const int wid  = tid >> 5;
    const int lane = tid & 31;
    const int laneM   = lane & 15;
    const int halfSel = lane >> 4;            // 0: lanes 0-15, 1: lanes 16-31

    const int mb = blockIdx.x * 64;
    const int nb = blockIdx.y * 128;
    const int kPerZ = K / gridDim.z;
    const int kb0   = blockIdx.z * kPerZ;

    const int wm = (wid & 1) * 32;            // 2 m-tiles (wm, wm+16)
    const int wn = (wid >> 1) * 32;           // 2 n-tiles (wn, wn+16)

    v8f acc[2][2] = {};

    // stage one 64x32 A tile + 32x128 B tile into LDS buffer `buf`
    auto stage = [&](int kk, int buf) {
        bf16_t* Asb = &As[buf * 64 * 40];
        bf16_t* Bsb = &Bs[buf * 128 * 40];
        // ---- A tile ----
        if (!AF32) {
            const bf16_t* A = (const bf16_t*)Av;
            int r = tid >> 2, c8 = (tid & 3) * 8;
            const bf16_t* gp = A + (size_t)(mb + r) * lda + kk + c8;
#if USE_ASYNC_LDS
            async_load_b128(gp, &Asb[r * 40 + c8]);
#else
            *(v8bf*)&Asb[r * 40 + c8] = *(const v8bf*)gp;
#endif
        } else if (!TRANSA) {
            const float* A = (const float*)Av;
            int r = tid >> 2, c = (tid & 3) * 8;
            const float4* p = (const float4*)(A + (size_t)(mb + r) * lda + kk + c);
            float4 f0 = p[0], f1 = p[1];
            __builtin_prefetch((const void*)(p + 16), 0, 0);   // k-tile two ahead (+64 floats) of u
            v8bf va;
            va[0] = (bf16_t)f0.x; va[1] = (bf16_t)f0.y; va[2] = (bf16_t)f0.z; va[3] = (bf16_t)f0.w;
            va[4] = (bf16_t)f1.x; va[5] = (bf16_t)f1.y; va[6] = (bf16_t)f1.z; va[7] = (bf16_t)f1.w;
            *(v8bf*)&Asb[r * 40 + c] = va;
        } else {
            const float* A = (const float*)Av;
#pragma unroll
            for (int i = 0; i < 2; i++) {
                int idx = tid + i * 256;
                int kr = idx >> 4, c4 = (idx & 15) * 4;
                const float4* p = (const float4*)(A + (size_t)(kk + kr) * lda + mb + c4);
                float4 f = *p;
                __builtin_prefetch((const void*)(p + 8 * lda), 0, 0);
                Asb[(c4 + 0) * 40 + kr] = (bf16_t)f.x;
                Asb[(c4 + 1) * 40 + kr] = (bf16_t)f.y;
                Asb[(c4 + 2) * 40 + kr] = (bf16_t)f.z;
                Asb[(c4 + 3) * 40 + kr] = (bf16_t)f.w;
            }
        }
        // ---- B tile (128 cols x 32 k, from Bt[N,K]) ----
        {
            int n = tid >> 1, c16 = (tid & 1) * 16;
            const bf16_t* bp = Bt + (size_t)(nb + n) * K + kk + c16;
#if USE_ASYNC_LDS
            async_load_b128(bp,     &Bsb[n * 40 + c16]);
            async_load_b128(bp + 8, &Bsb[n * 40 + c16 + 8]);
#else
            *(v8bf*)&Bsb[n * 40 + c16]     = *(const v8bf*)bp;
            *(v8bf*)&Bsb[n * 40 + c16 + 8] = *(const v8bf*)(bp + 8);
#endif
        }
    };

    const int nk = kPerZ / 32;
    stage(kb0, 0);

    for (int t = 0; t < nk; t++) {
#if USE_ASYNC_LDS
        wait_async0();
#endif
        __syncthreads();                       // buffer t&1 ready; prev reads done
        if (t + 1 < nk) stage(kb0 + (t + 1) * 32, (t + 1) & 1);

        const bf16_t* Asb = &As[(t & 1) * 64 * 40];
        const bf16_t* Bsb = &Bs[(t & 1) * 128 * 40];

        // A fragments (16x32): lanes 0-15 K {0..7,16..23}, lanes 16-31 K {8..15,24..31}
        v16bf afrag[2];
#pragma unroll
        for (int mi = 0; mi < 2; mi++) {
            int arow  = wm + mi * 16 + laneM;
            int abase = halfSel * 8;
            v8bf alo = *(const v8bf*)&Asb[arow * 40 + abase];
            v8bf ahi = *(const v8bf*)&Asb[arow * 40 + abase + 16];
            afrag[mi] = __builtin_shufflevector(alo, ahi, 0, 1, 2, 3, 4, 5, 6, 7,
                                                8, 9, 10, 11, 12, 13, 14, 15);
        }
        // B fragments (32x16): lane = column, lanes 0-15 K {0..7,16..23}, lanes 16-31 K {8..15,24..31}
        v16bf bfrag[2];
#pragma unroll
        for (int ni = 0; ni < 2; ni++) {
            int bcol = wn + ni * 16 + laneM;
            int kb2  = halfSel * 8;
            v8bf blo = *(const v8bf*)&Bsb[bcol * 40 + kb2];
            v8bf bhi = *(const v8bf*)&Bsb[bcol * 40 + kb2 + 16];
            bfrag[ni] = __builtin_shufflevector(blo, bhi, 0, 1, 2, 3, 4, 5, 6, 7,
                                                8, 9, 10, 11, 12, 13, 14, 15);
        }
#pragma unroll
        for (int mi = 0; mi < 2; mi++)
#pragma unroll
            for (int ni = 0; ni < 2; ni++)
                acc[mi][ni] = __builtin_amdgcn_wmma_f32_16x16x32_bf16(
                    false, afrag[mi], false, bfrag[ni], (short)0, acc[mi][ni], false, false);
    }

    // ---- epilogue: C 16x16 f32: VGPR r, lanes 0-15 -> M=r, lanes 16-31 -> M=r+8
#pragma unroll
    for (int mi = 0; mi < 2; mi++) {
        const int rowb = mb + wm + mi * 16 + halfSel * 8;
#pragma unroll
        for (int ni = 0; ni < 2; ni++) {
            int colg = nb + wn + ni * 16 + laneM;
            float bv = (bias != nullptr) ? bias[colg] : 0.0f;
#pragma unroll
            for (int r = 0; r < 8; r++) {
                float v = acc[mi][ni][r] + bv;
                if (ACT == 1) v = fmaxf(v, 0.0f);
                else if (ACT == 2) v = gelu_exact(v);
                size_t idx = (size_t)(rowb + r) * N + colg;
                if (ATOMIC) {
                    atomicAdd(Cf + idx, v);
                } else {
                    if (Cf)  Cf[idx]  = v;
                    if (Cbf) Cbf[idx] = (bf16_t)v;
                }
            }
        }
    }
}

// ---------------- small kernels ----------------
__global__ void cvt_bf16_kernel(const float* __restrict__ src, bf16_t* __restrict__ dst, int n) {
    int i = blockIdx.x * blockDim.x + threadIdx.x;
    if (i < n) dst[i] = (bf16_t)src[i];
}

// dst[c*rows + r] = src[r*cols + c]   (dst = src^T as bf16, [cols x rows])
__global__ void transcvt_kernel(const float* __restrict__ src, bf16_t* __restrict__ dst,
                                int rows, int cols) {
    int idx = blockIdx.x * blockDim.x + threadIdx.x;
    int total = rows * cols;
    if (idx >= total) return;
    int c = idx / rows;
    int r = idx - c * rows;
    dst[idx] = (bf16_t)src[(size_t)r * cols + c];
}

__global__ void zero_kernel(float* __restrict__ p, int n) {
    int i = blockIdx.x * blockDim.x + threadIdx.x;
    if (i < n) p[i] = 0.0f;
}

__global__ void add3_kernel(const float* a, const float* b, const float* c,
                            float* __restrict__ o, int n) {
    int i = blockIdx.x * blockDim.x + threadIdx.x;
    if (i < n) o[i] = a[i] + b[i] + c[i];
}

__global__ void add2_kernel(const float* a, const float* b, float* __restrict__ o, int n) {
    int i = blockIdx.x * blockDim.x + threadIdx.x;
    if (i < n) o[i] = a[i] + b[i];
}

// SineEncoding -> new_e[n]
__global__ void sine_kernel(const float* __restrict__ e, const float* __restrict__ eig_w,
                            const float* __restrict__ eig_b, const float* __restrict__ alpha_w,
                            float* __restrict__ newe) {
    int n = blockIdx.x * blockDim.x + threadIdx.x;
    if (n >= NN) return;
    float ev = e[n];
    float p = 1.0f;
    float acc = 0.0f;
#pragma unroll 1
    for (int k = 0; k < KPOW; k++) {
        p *= ev;                                   // e^(k+1)
        float o = eig_w[k * 33 + 0];               // ones term
#pragma unroll
        for (int d = 1; d <= DIMH; d++) {
            float sv, cv;
            sincosf(p * (float)d, &sv, &cv);
            o += eig_w[k * 33 + d] * sv + eig_w[k * 33 + DIMH + d] * cv;
        }
        o += eig_b[k];
        acc += o * alpha_w[k];
    }
    newe[n] = acc;
}

// LayerNorm over C=128, one wave per row, bf16 out
__global__ __launch_bounds__(256)
void ln_kernel(const float* __restrict__ src, const float* __restrict__ g,
               const float* __restrict__ b, bf16_t* __restrict__ dst, int rows) {
    int wid = threadIdx.x >> 5, lane = threadIdx.x & 31;
    int row = blockIdx.x * 8 + wid;
    if (row >= rows) return;
    const float* p = src + (size_t)row * CC;
    float v[4], s = 0.0f, sq = 0.0f;
#pragma unroll
    for (int i = 0; i < 4; i++) {
        v[i] = p[lane + i * 32];
        s += v[i]; sq += v[i] * v[i];
    }
#pragma unroll
    for (int off = 16; off >= 1; off >>= 1) {
        s  += __shfl_xor(s,  off, 32);
        sq += __shfl_xor(sq, off, 32);
    }
    float m   = s * (1.0f / 128.0f);
    float var = sq * (1.0f / 128.0f) - m * m;
    float inv = rsqrtf(var + 1e-5f);
#pragma unroll
    for (int i = 0; i < 4; i++) {
        int c = lane + i * 32;
        dst[(size_t)row * CC + c] = (bf16_t)((v[i] - m) * inv * g[c] + b[c]);
    }
}

// Pzt[c][n] = bf16(new_e[n] * P[n][c])   (transposed for B-operand use)
__global__ void scale_pzt_kernel(const float* __restrict__ P, const float* __restrict__ newe,
                                 bf16_t* __restrict__ dst) {
    int idx = blockIdx.x * blockDim.x + threadIdx.x;
    if (idx >= CC * NN) return;
    int c = idx >> 13;          // / 8192
    int n = idx & (NN - 1);
    dst[idx] = (bf16_t)(P[(size_t)n * CC + c] * newe[n]);
}

// ---------------- launcher ----------------
extern "C" void kernel_launch(void* const* d_in, const int* in_sizes, int n_in,
                              void* d_out, int out_size, void* d_ws, size_t ws_size,
                              hipStream_t stream) {
    (void)in_sizes; (void)n_in; (void)out_size; (void)ws_size;
    const float* e      = (const float*)d_in[0];
    const float* u      = (const float*)d_in[1];
    const float* x      = (const float*)d_in[2];
    const float* fe_w1  = (const float*)d_in[3];
    const float* fe_b1  = (const float*)d_in[4];
    const float* fe_w2  = (const float*)d_in[5];
    const float* fe_b2  = (const float*)d_in[6];
    const float* eig_w  = (const float*)d_in[7];
    const float* eig_b  = (const float*)d_in[8];
    const float* alphaw = (const float*)d_in[9];
    const float* mha_g  = (const float*)d_in[10];
    const float* mha_b  = (const float*)d_in[11];
    const float* ffn_g  = (const float*)d_in[12];
    const float* ffn_b  = (const float*)d_in[13];
    const float* wq = (const float*)d_in[14]; const float* bq = (const float*)d_in[15];
    const float* wk = (const float*)d_in[16]; const float* bk = (const float*)d_in[17];
    const float* wv = (const float*)d_in[18]; const float* bv = (const float*)d_in[19];
    const float* wo = (const float*)d_in[20]; const float* bo = (const float*)d_in[21];
    const float* f1w = (const float*)d_in[22]; const float* f1b = (const float*)d_in[23];
    const float* f2w = (const float*)d_in[24]; const float* f2b = (const float*)d_in[25];
    float* out = (float*)d_out;

    // workspace carve-up (256B aligned)
    size_t off = 0;
    auto wsa = [&](size_t bytes) -> void* {
        void* p = (char*)d_ws + off;
        off += (bytes + 255) & ~(size_t)255;
        return p;
    };
    bf16_t* x_bf   = (bf16_t*)wsa((size_t)NN * NFEAT * 2);
    bf16_t* w1t    = (bf16_t*)wsa((size_t)NFEAT * HID * 2);
    bf16_t* w2t    = (bf16_t*)wsa((size_t)HID * CC * 2);
    bf16_t* wqt    = (bf16_t*)wsa((size_t)CC * CC * 2);
    bf16_t* wkt    = (bf16_t*)wsa((size_t)CC * CC * 2);
    bf16_t* wvt    = (bf16_t*)wsa((size_t)CC * CC * 2);
    bf16_t* wot    = (bf16_t*)wsa((size_t)CC * CC * 2);
    bf16_t* f1t    = (bf16_t*)wsa((size_t)CC * CC * 2);
    bf16_t* f2t    = (bf16_t*)wsa((size_t)CC * CC * 2);
    bf16_t* hid_bf = (bf16_t*)wsa((size_t)NN * HID * 2);
    float*  h      = (float*) wsa((size_t)NN * CC * 4);
    bf16_t* ht_bf  = (bf16_t*)wsa((size_t)NN * CC * 2);
    float*  newe   = (float*) wsa((size_t)NN * 4);
    float*  P      = (float*) wsa((size_t)NN * CC * 4);
    bf16_t* pzt    = (bf16_t*)wsa((size_t)NN * CC * 2);
    float*  hfur   = (float*) wsa((size_t)NN * CC * 4);
    bf16_t* mh_bf  = (bf16_t*)wsa((size_t)NN * CC * 2);
    bf16_t* q_bf   = (bf16_t*)wsa((size_t)NN * CC * 2);
    float*  kf     = (float*) wsa((size_t)NN * CC * 4);
    float*  vf     = (float*) wsa((size_t)NN * CC * 4);
    bf16_t* vt_bf  = (bf16_t*)wsa((size_t)NN * CC * 2);
    float*  ktv    = (float*) wsa((size_t)CC * CC * 4);
    bf16_t* ktvt   = (bf16_t*)wsa((size_t)CC * CC * 2);
    bf16_t* att_bf = (bf16_t*)wsa((size_t)NN * CC * 2);
    float*  att2   = (float*) wsa((size_t)NN * CC * 4);
    float*  h1     = (float*) wsa((size_t)NN * CC * 4);
    bf16_t* fln_bf = (bf16_t*)wsa((size_t)NN * CC * 2);
    bf16_t* f1_bf  = (bf16_t*)wsa((size_t)NN * CC * 2);
    float*  f2buf  = (float*) wsa((size_t)NN * CC * 4);

    const int T = 256;
    auto blocks = [](int n) { return (n + 255) / 256; };

    // --- operand conversions (tiny, L2 resident) ---
    cvt_bf16_kernel<<<blocks(NN * NFEAT), T, 0, stream>>>(x, x_bf, NN * NFEAT);
    transcvt_kernel<<<blocks(NFEAT * HID), T, 0, stream>>>(fe_w1, w1t, NFEAT, HID);
    transcvt_kernel<<<blocks(HID * CC), T, 0, stream>>>(fe_w2, w2t, HID, CC);
    transcvt_kernel<<<blocks(CC * CC), T, 0, stream>>>(wq, wqt, CC, CC);
    transcvt_kernel<<<blocks(CC * CC), T, 0, stream>>>(wk, wkt, CC, CC);
    transcvt_kernel<<<blocks(CC * CC), T, 0, stream>>>(wv, wvt, CC, CC);
    transcvt_kernel<<<blocks(CC * CC), T, 0, stream>>>(wo, wot, CC, CC);
    transcvt_kernel<<<blocks(CC * CC), T, 0, stream>>>(f1w, f1t, CC, CC);
    transcvt_kernel<<<blocks(CC * CC), T, 0, stream>>>(f2w, f2t, CC, CC);

    // --- sine encoding -> new_e ---
    sine_kernel<<<blocks(NN), T, 0, stream>>>(e, eig_w, eig_b, alphaw, newe);

    // --- feature encoder ---
    gemm_wmma<1, false, false, false><<<dim3(NN / 64, HID / 128, 1), T, 0, stream>>>(
        x_bf, w1t, fe_b1, nullptr, hid_bf, NN, HID, NFEAT, NFEAT);
    gemm_wmma<0, false, false, false><<<dim3(NN / 64, CC / 128, 1), T, 0, stream>>>(
        hid_bf, w2t, fe_b2, h, nullptr, NN, CC, HID, HID);
    transcvt_kernel<<<blocks(NN * CC), T, 0, stream>>>(h, ht_bf, NN, CC);

    // --- spectral propagation: P = u^T @ h ; h_fur = u @ (new_e .* P) ---
    gemm_wmma<0, true, true, false><<<dim3(NN / 64, CC / 128, 1), T, 0, stream>>>(
        u, ht_bf, nullptr, P, nullptr, NN, CC, NN, NN);
    scale_pzt_kernel<<<blocks(NN * CC), T, 0, stream>>>(P, newe, pzt);
    gemm_wmma<0, false, true, false><<<dim3(NN / 64, CC / 128, 1), T, 0, stream>>>(
        u, pzt, nullptr, hfur, nullptr, NN, CC, NN, NN);

    // --- linear attention ---
    ln_kernel<<<NN / 8, T, 0, stream>>>(h, mha_g, mha_b, mh_bf, NN);
    gemm_wmma<0, false, false, false><<<dim3(NN / 64, 1, 1), T, 0, stream>>>(
        mh_bf, wqt, bq, nullptr, q_bf, NN, CC, CC, CC);
    gemm_wmma<0, false, false, false><<<dim3(NN / 64, 1, 1), T, 0, stream>>>(
        mh_bf, wkt, bk, kf, nullptr, NN, CC, CC, CC);
    gemm_wmma<0, false, false, false><<<dim3(NN / 64, 1, 1), T, 0, stream>>>(
        mh_bf, wvt, bv, vf, nullptr, NN, CC, CC, CC);
    transcvt_kernel<<<blocks(NN * CC), T, 0, stream>>>(vf, vt_bf, NN, CC);

    // kTv = k^T @ v : split-K over 64 chunks, atomic accumulate
    zero_kernel<<<blocks(CC * CC), T, 0, stream>>>(ktv, CC * CC);
    gemm_wmma<0, true, true, true><<<dim3(CC / 64, 1, 64), T, 0, stream>>>(
        kf, vt_bf, nullptr, ktv, nullptr, CC, CC, NN, CC);
    transcvt_kernel<<<blocks(CC * CC), T, 0, stream>>>(ktv, ktvt, CC, CC);

    // att = (q @ kTv) @ wo + bo
    gemm_wmma<0, false, false, false><<<dim3(NN / 64, 1, 1), T, 0, stream>>>(
        q_bf, ktvt, nullptr, nullptr, att_bf, NN, CC, CC, CC);
    gemm_wmma<0, false, false, false><<<dim3(NN / 64, 1, 1), T, 0, stream>>>(
        att_bf, wot, bo, att2, nullptr, NN, CC, CC, CC);

    // h1 = h + att + h_fur
    add3_kernel<<<blocks(NN * CC), T, 0, stream>>>(h, att2, hfur, h1, NN * CC);

    // --- FFN ---
    ln_kernel<<<NN / 8, T, 0, stream>>>(h1, ffn_g, ffn_b, fln_bf, NN);
    gemm_wmma<2, false, false, false><<<dim3(NN / 64, 1, 1), T, 0, stream>>>(
        fln_bf, f1t, f1b, nullptr, f1_bf, NN, CC, CC, CC);
    gemm_wmma<0, false, false, false><<<dim3(NN / 64, 1, 1), T, 0, stream>>>(
        f1_bf, f2t, f2b, f2buf, nullptr, NN, CC, CC, CC);

    // out = h1 + f
    add2_kernel<<<blocks(NN * CC), T, 0, stream>>>(h1, f2buf, out, NN * CC);
}
